// S5Layer_38285338476550
// MI455X (gfx1250) — matrix operand order repaired and checked by reading, not compiled, not run on hardware
//
#include <hip/hip_runtime.h>
#include <hip/hip_bf16.h>
#include <stdint.h>

typedef __bf16 bf16;
typedef __attribute__((ext_vector_type(16))) __bf16 v16bf;
typedef __attribute__((ext_vector_type(8)))  __bf16 v8bf;
typedef __attribute__((ext_vector_type(8)))  float   v8f;
typedef uint32_t u32;
typedef __attribute__((ext_vector_type(4))) u32 u32x4;
typedef __attribute__((ext_vector_type(8))) int i32x8;
typedef __attribute__((ext_vector_type(4))) int i32x4;

#define Bsz   4
#define Lsz   4096
#define Hsz   512
#define Nsz   128
#define Msz   (Bsz*Lsz)        /* 16384 rows */
#define CHUNK 256
#define NCH   (Lsz/CHUNK)      /* 16 chunks per batch */
#define TOTCH (Bsz*NCH)        /* 64 */

/* workspace layout (float offsets) */
#define WS_BU_R  ((size_t)0)
#define WS_BU_I  ((size_t)Msz*(size_t)Nsz)                 /* 2,097,152 */
#define WS_COEFF (WS_BU_I + (size_t)Msz*Nsz)               /* 4,194,304 */
#define WS_POW   (WS_COEFF + 2*Nsz)                        /* Lbar^k, k=0..CHUNK */
#define WS_E     (WS_POW + (size_t)(CHUNK+1)*Nsz*2)
#define WS_CARRY (WS_E + (size_t)TOTCH*Nsz*2)
#define WS_BF16F (WS_CARRY + (size_t)TOTCH*Nsz*2)          /* bf16 region base */
#define OUT_MAIN ((size_t)Msz*Hsz)                         /* 8,388,608 */

/* bf16 offsets inside the bf16 region */
#define BF_BR ((size_t)0)
#define BF_BI ((size_t)(Nsz*Hsz))          /* 65536  */
#define BF_CR ((size_t)(2*Nsz*Hsz))        /* 131072 */
#define BF_CI ((size_t)(3*Nsz*Hsz))        /* 196608 */
#define BF_W  ((size_t)(4*Nsz*Hsz))        /* 262144 */
#define BF_TOT ((size_t)(4*Nsz*Hsz + Hsz*Hsz))   /* 524288 */

/* ---------------- WMMA helpers (bf16 in, f32 accumulate) ---------------- */

__device__ __forceinline__ v8f wmma_bf16(v16bf a, v16bf b, v8f c) {
  return __builtin_amdgcn_wmma_f32_16x16x32_bf16(false, a, false, b, (short)0, c,
                                                 false, false);
}

/* A-matrix 16x32 bf16 layout: lanes 0-15 -> M=lane, elems[0..7]=K k0+0..7,
   elems[8..15]=K k0+16..23; lanes 16-31 -> M=lane-16, K offsets +8 / +24. */
__device__ __forceinline__ v16bf load_a_g(const float* __restrict__ base, int ld,
                                          int row, int k0, int lane, float scale) {
  const float* p = base + (size_t)row * ld + k0 + ((lane >> 4) << 3);
  float4 v0 = *(const float4*)(p);
  float4 v1 = *(const float4*)(p + 4);
  float4 v2 = *(const float4*)(p + 16);
  float4 v3 = *(const float4*)(p + 20);
  v16bf a;
  a[0]=(bf16)(v0.x*scale); a[1]=(bf16)(v0.y*scale); a[2]=(bf16)(v0.z*scale); a[3]=(bf16)(v0.w*scale);
  a[4]=(bf16)(v1.x*scale); a[5]=(bf16)(v1.y*scale); a[6]=(bf16)(v1.z*scale); a[7]=(bf16)(v1.w*scale);
  a[8]=(bf16)(v2.x*scale); a[9]=(bf16)(v2.y*scale); a[10]=(bf16)(v2.z*scale); a[11]=(bf16)(v2.w*scale);
  a[12]=(bf16)(v3.x*scale); a[13]=(bf16)(v3.y*scale); a[14]=(bf16)(v3.z*scale); a[15]=(bf16)(v3.w*scale);
  return a;
}

/* B-matrix 32x16: lanes 0-15 -> N=lane, K=k0+0..7,16..23; lanes 16-31 -> N=lane-16,
   K=k0+8..15,24..31. Direct bf16 source (pre-converted), 16B-aligned vector loads. */
__device__ __forceinline__ v16bf load_b_bf(const bf16* __restrict__ base, int ld,
                                           int nrow, int k0, int lane) {
  const bf16* p = base + (size_t)nrow * ld + k0 + ((lane >> 4) << 3);
  v8bf lo = *(const v8bf*)(p);
  v8bf hi = *(const v8bf*)(p + 16);
  v16bf b;
#pragma unroll
  for (int i = 0; i < 8; ++i) { b[i] = lo[i]; b[8 + i] = hi[i]; }
  return b;
}

/* A operand from fp32 LDS tile (TDM-staged), cvt on read */
__device__ __forceinline__ v16bf load_a_lds_f32(const float* __restrict__ p) {
  v16bf a;
#pragma unroll
  for (int i = 0; i < 8; ++i) a[i] = (bf16)p[i];
#pragma unroll
  for (int i = 0; i < 8; ++i) a[8 + i] = (bf16)p[16 + i];
  return a;
}

__device__ __forceinline__ void zero_acc(v8f* acc, int n) {
  for (int t = 0; t < n; ++t)
#pragma unroll
    for (int i = 0; i < 8; ++i) acc[t][i] = 0.f;
}

/* ---- TDM: 2D tile (fp32 elements) global -> LDS, one issue per block ---- */
__device__ __forceinline__ void tdm_load_2d_f32(u32 lds_addr, const void* gptr,
                                                u32 dim0, u32 dim1,
                                                u32 tdim0, u32 tdim1, u32 stride0) {
  uint64_t ga = (uint64_t)(uintptr_t)gptr;
  u32x4 g0;
  g0[0] = 1u;                                         /* count=1, user D#      */
  g0[1] = lds_addr;                                   /* lds_addr [63:32]      */
  g0[2] = (u32)(ga & 0xffffffffu);                    /* global_addr lo        */
  g0[3] = (u32)((ga >> 32) & 0x01ffffffu) | (2u << 30); /* ga hi | type=2      */
  i32x8 g1;
  g1[0] = (int)(2u << 16);                            /* data_size=4B, mask=0  */
  g1[1] = (int)((dim0 & 0xffffu) << 16);              /* tensor_dim0 lo16      */
  g1[2] = (int)((dim0 >> 16) | ((dim1 & 0xffffu) << 16)); /* d0 hi | d1 lo     */
  g1[3] = (int)((dim1 >> 16) | (tdim0 << 16));        /* d1 hi | tile_dim0     */
  g1[4] = (int)(tdim1);                               /* tile_dim1, tile_dim2=0*/
  g1[5] = (int)(stride0);                             /* tensor_dim0_stride lo */
  g1[6] = 0;
  g1[7] = 0;
  i32x4 z4 = {0, 0, 0, 0};
#if __has_include(<hip/amd_detail/amd_gfx1250_TDM.h>)
  i32x8 z8 = {0, 0, 0, 0, 0, 0, 0, 0};
  __builtin_amdgcn_tensor_load_to_lds(g0, g1, z4, z4, z8, 0);
#else
  __builtin_amdgcn_tensor_load_to_lds(g0, g1, z4, z4, 0);
#endif
}

/* ---------------- 0) pre-convert static matrices to bf16 ---------------- */
__global__ void __launch_bounds__(256) k_cvt(const float* __restrict__ Br,
                                             const float* __restrict__ Bi,
                                             const float* __restrict__ Cr,
                                             const float* __restrict__ Ci,
                                             const float* __restrict__ W,
                                             bf16* __restrict__ dst) {
  size_t i = (size_t)blockIdx.x * 256 + threadIdx.x;   /* 0..BF_TOT-1 */
  float v;
  if (i < BF_BI)      v = Br[i - BF_BR];
  else if (i < BF_CR) v = Bi[i - BF_BI];
  else if (i < BF_CI) v = Cr[i - BF_CR];
  else if (i < BF_W)  v = Ci[i - BF_CI];
  else                v = W[i - BF_W];
  dst[i] = (bf16)v;
}

/* ---------------- 1) coefficients + decay powers ---------------- */
__global__ void __launch_bounds__(128) k_coeff(const float* __restrict__ log_lam_r,
                                               const float* __restrict__ lam_i,
                                               const float* __restrict__ log_dt,
                                               float* __restrict__ ws) {
  int n = threadIdx.x;
  float dt = __expf(log_dt[0]);
  float lamr = -__expf(log_lam_r[n]);
  float decay = __expf(lamr * dt);
  float lr = decay * __cosf(lam_i[n] * dt);
  float li = decay * __sinf(lam_i[n] * dt);
  ws[WS_COEFF + n] = lr;
  ws[WS_COEFF + Nsz + n] = li;
  float* Pr = ws + WS_POW;
  float* Pi = ws + WS_POW + (size_t)(CHUNK + 1) * Nsz;
  float pr = 1.f, pi = 0.f;
  Pr[n] = pr; Pi[n] = pi;
  for (int k = 1; k <= CHUNK; ++k) {
    float nr = pr * lr - pi * li;
    float ni = pr * li + pi * lr;
    pr = nr; pi = ni;
    Pr[(size_t)k * Nsz + n] = pr;
    Pi[(size_t)k * Nsz + n] = pi;
  }
}

/* ---------------- 2) Bu = dt * x @ B^T (real & imag) ---------------- */
__global__ void __launch_bounds__(256) k_gemm_bu(const float* __restrict__ x,
                                                 const bf16* __restrict__ wb,
                                                 const float* __restrict__ log_dt,
                                                 float* __restrict__ ws) {
  const int lane = threadIdx.x & 31;
  const int wave = threadIdx.x >> 5;
  const int wm = wave & 3;       /* 4 M-tiles of 16 rows */
  const int wi = wave >> 2;      /* 0 -> real, 1 -> imag */
  const int m0 = blockIdx.x * 64 + wm * 16;
  const float dt = __expf(log_dt[0]);
  const bf16* Bsrc = wb + (wi ? BF_BI : BF_BR);
  float* Out = ws + (wi ? WS_BU_I : WS_BU_R);

  v8f acc[8];
  zero_acc(acc, 8);
  const int arow = m0 + (lane & 15);
  const int nb = lane & 15;
  for (int k0 = 0; k0 < Hsz; k0 += 32) {
    __builtin_prefetch(x + (size_t)arow * Hsz + k0 + 64, 0, 1);
    v16bf a = load_a_g(x, Hsz, arow, k0, lane, dt);
#pragma unroll
    for (int t = 0; t < 8; ++t) {
      v16bf b = load_b_bf(Bsrc, Hsz, t * 16 + nb, k0, lane);
      acc[t] = wmma_bf16(a, b, acc[t]);
    }
  }
  const int mrb = m0 + ((lane >> 4) << 3);
#pragma unroll
  for (int t = 0; t < 8; ++t) {
    int n = t * 16 + nb;
#pragma unroll
    for (int r = 0; r < 8; ++r)
      Out[(size_t)(mrb + r) * Nsz + n] = acc[t][r];
  }
}

/* ---------------- 3) local scan within each chunk (in place) ------------- */
__global__ void __launch_bounds__(128) k_scan_local(float* __restrict__ ws) {
  const int blk = blockIdx.x;           /* 0..63 */
  const int b = blk / NCH, c = blk % NCH;
  const int n = threadIdx.x;
  const float lr = ws[WS_COEFF + n];
  const float li = ws[WS_COEFF + Nsz + n];
  float* Sr = ws + WS_BU_R;
  float* Si = ws + WS_BU_I;
  size_t base = ((size_t)b * Lsz + (size_t)c * CHUNK) * Nsz + n;
  float sr = 0.f, si = 0.f;
  for (int t = 0; t < CHUNK; ++t) {
    size_t idx = base + (size_t)t * Nsz;
    float br = Sr[idx], bi = Si[idx];
    float nr = lr * sr - li * si + br;
    float ni = lr * si + li * sr + bi;
    sr = nr; si = ni;
    Sr[idx] = sr; Si[idx] = si;
  }
  ws[WS_E + (size_t)blk * Nsz + n] = sr;
  ws[WS_E + (size_t)TOTCH * Nsz + (size_t)blk * Nsz + n] = si;
}

/* ---------------- 4) serial carry across chunks + final state ----------- */
__global__ void __launch_bounds__(128) k_scan_carry(float* __restrict__ ws,
                                                    float* __restrict__ out) {
  const int b = blockIdx.x;             /* 0..3 */
  const int n = threadIdx.x;
  const float Ar = ws[WS_POW + (size_t)CHUNK * Nsz + n];
  const float Ai = ws[WS_POW + (size_t)(CHUNK + 1) * Nsz + (size_t)CHUNK * Nsz + n];
  float cr = 0.f, ci = 0.f;
  for (int c = 0; c < NCH; ++c) {
    size_t e = (size_t)(b * NCH + c) * Nsz + n;
    ws[WS_CARRY + e] = cr;
    ws[WS_CARRY + (size_t)TOTCH * Nsz + e] = ci;
    float er = ws[WS_E + e];
    float ei = ws[WS_E + (size_t)TOTCH * Nsz + e];
    float nr = Ar * cr - Ai * ci + er;
    float ni = Ar * ci + Ai * cr + ei;
    cr = nr; ci = ni;
  }
  out[OUT_MAIN + (size_t)b * Nsz + n] = cr;               /* sr_f */
  out[OUT_MAIN + Bsz * Nsz + (size_t)b * Nsz + n] = ci;   /* si_f */
}

/* ---------------- 5) fix-up: S[t] += Lbar^(tl+1) * carry ---------------- */
__global__ void __launch_bounds__(256) k_scan_fix(float* __restrict__ ws) {
  size_t e = (size_t)blockIdx.x * 256 + threadIdx.x;      /* complex elem id */
  int n = (int)(e & (Nsz - 1));
  size_t mt = e >> 7;                                     /* b*L + t */
  int t = (int)(mt & (Lsz - 1));
  int b = (int)(mt >> 12);
  int tl = t & (CHUNK - 1);
  int c = t >> 8;
  size_t ce = (size_t)(b * NCH + c) * Nsz + n;
  float cr = ws[WS_CARRY + ce];
  float ci = ws[WS_CARRY + (size_t)TOTCH * Nsz + ce];
  float pr = ws[WS_POW + (size_t)(tl + 1) * Nsz + n];
  float pi = ws[WS_POW + (size_t)(CHUNK + 1) * Nsz + (size_t)(tl + 1) * Nsz + n];
  ws[WS_BU_R + e] += pr * cr - pi * ci;
  ws[WS_BU_I + e] += pr * ci + pi * cr;
}

/* ---------------- 6) y = Sr@Cr^T - Si@Ci^T + D*x  (-> d_out staging) ---- */
__global__ void __launch_bounds__(256) k_gemm_y(const float* __restrict__ ws,
                                                const bf16* __restrict__ wb,
                                                const float* __restrict__ Dv,
                                                const float* __restrict__ x,
                                                float* __restrict__ y) {
  const int lane = threadIdx.x & 31;
  const int wave = threadIdx.x >> 5;
  const int wm = wave & 3;
  const int wg = wave >> 2;
  const int m0 = blockIdx.x * 64 + wm * 16;
  const int h0 = blockIdx.y * 256 + wg * 128;
  const float* Sr = ws + WS_BU_R;
  const float* Si = ws + WS_BU_I;
  const bf16* Crb = wb + BF_CR;
  const bf16* Cib = wb + BF_CI;

  v8f acc[8];
  zero_acc(acc, 8);
  const int arow = m0 + (lane & 15);
  const int nb = lane & 15;
  for (int k0 = 0; k0 < Nsz; k0 += 32) {
    v16bf ar = load_a_g(Sr, Nsz, arow, k0, lane, 1.f);
    v16bf ai = load_a_g(Si, Nsz, arow, k0, lane, -1.f);   /* minus sign folded */
#pragma unroll
    for (int t = 0; t < 8; ++t) {
      int h = h0 + t * 16 + nb;
      acc[t] = wmma_bf16(ar, load_b_bf(Crb, Nsz, h, k0, lane), acc[t]);
      acc[t] = wmma_bf16(ai, load_b_bf(Cib, Nsz, h, k0, lane), acc[t]);
    }
  }
  const int mrb = m0 + ((lane >> 4) << 3);
#pragma unroll
  for (int t = 0; t < 8; ++t) {
    int h = h0 + t * 16 + nb;
    float d = Dv[h];
#pragma unroll
    for (int r = 0; r < 8; ++r) {
      size_t idx = (size_t)(mrb + r) * Hsz + h;
      y[idx] = acc[t][r] + d * x[idx];
    }
  }
}

/* ------- 7) out = y @ W_out^T + b_out (in-place, TDM-staged LDS) -------- */
__global__ void __launch_bounds__(256) k_gemm_out(float* __restrict__ out,
                                                  const bf16* __restrict__ wb,
                                                  const float* __restrict__ bo) {
  __shared__ float ysf[64][512];        /* 128KB; matches TDM contiguous tile */
  const int m0 = blockIdx.x * 64;
  const int lane = threadIdx.x & 31;
  const int wave = threadIdx.x >> 5;

  if (wave == 0) {                      /* one TDM issue per block */
    tdm_load_2d_f32((u32)(uintptr_t)&ysf[0][0], out + (size_t)m0 * Hsz,
                    /*dim0*/ Hsz, /*dim1*/ 64, /*tile0*/ Hsz, /*tile1*/ 64,
                    /*stride0*/ Hsz);
    __builtin_amdgcn_s_wait_tensorcnt(0);
  }
  __syncthreads();

  const bf16* W = wb + BF_W;
  const int wm = wave & 3;
  const int wg = wave >> 2;             /* h half: 0..1, 256 wide each */
  const int h0 = wg * 256;
  v8f acc[16];
  zero_acc(acc, 16);
  const int lrow = wm * 16 + (lane & 15);
  const int nb = lane & 15;
  for (int k0 = 0; k0 < Hsz; k0 += 32) {
    __builtin_prefetch(W + (size_t)(h0 + nb) * Hsz + k0 + 64, 0, 1);
    v16bf a = load_a_lds_f32(&ysf[lrow][k0 + ((lane >> 4) << 3)]);
#pragma unroll
    for (int t = 0; t < 16; ++t) {
      v16bf b = load_b_bf(W, Hsz, h0 + t * 16 + nb, k0, lane);
      acc[t] = wmma_bf16(a, b, acc[t]);
    }
  }
  const int mrb = m0 + wm * 16 + ((lane >> 4) << 3);
#pragma unroll
  for (int t = 0; t < 16; ++t) {
    int h = h0 + t * 16 + nb;
    float bb = bo[h];
#pragma unroll
    for (int r = 0; r < 8; ++r)
      out[(size_t)(mrb + r) * Hsz + h] = acc[t][r] + bb;
  }
}

/* ---------------- host-side launch ---------------- */
extern "C" void kernel_launch(void* const* d_in, const int* in_sizes, int n_in,
                              void* d_out, int out_size, void* d_ws, size_t ws_size,
                              hipStream_t stream) {
  (void)in_sizes; (void)n_in; (void)out_size; (void)ws_size;
  const float* x   = (const float*)d_in[0];
  const float* llr = (const float*)d_in[1];
  const float* lim = (const float*)d_in[2];
  const float* Brm = (const float*)d_in[3];
  const float* Bim = (const float*)d_in[4];
  const float* Cr  = (const float*)d_in[5];
  const float* Ci  = (const float*)d_in[6];
  const float* Dv  = (const float*)d_in[7];
  const float* ldt = (const float*)d_in[8];
  const float* W   = (const float*)d_in[9];
  const float* bo  = (const float*)d_in[10];
  float* out = (float*)d_out;
  float* ws  = (float*)d_ws;
  bf16* wb   = (bf16*)(ws + WS_BF16F);

  k_coeff<<<1, 128, 0, stream>>>(llr, lim, ldt, ws);
  k_cvt<<<(int)(BF_TOT / 256), 256, 0, stream>>>(Brm, Bim, Cr, Ci, W, wb);
  k_gemm_bu<<<Msz / 64, 256, 0, stream>>>(x, wb, ldt, ws);
  k_scan_local<<<TOTCH, 128, 0, stream>>>(ws);
  k_scan_carry<<<Bsz, 128, 0, stream>>>(ws, out);
  k_scan_fix<<<(Msz * Nsz) / 256, 256, 0, stream>>>(ws);
  k_gemm_y<<<dim3(Msz / 64, 2), 256, 0, stream>>>(ws, wb, Dv, x, out);
  k_gemm_out<<<Msz / 64, 256, 0, stream>>>(out, wb, bo);
}
